// FastRCNNPredictorSNNFull_22522808500921
// MI455X (gfx1250) — compile-verified, matrix operand order checked by reading, not run on hardware
//
#include <hip/hip_runtime.h>
#include <hip/hip_bf16.h>
#include <cstdint>
#include <cstddef>

// ---------------------------------------------------------------------------
// Types for CDNA5 WMMA (wave32): V_WMMA_F32_16X16X32_BF16
// ---------------------------------------------------------------------------
typedef __attribute__((ext_vector_type(16))) __bf16 v16bf;
typedef __attribute__((ext_vector_type(8)))  float  v8f;

union FragU { uint4 u4[2]; v16bf v; };   // 32 bytes: one 16x32 bf16 A/B fragment

#define DTM     0.1f    // DT * TAU_MEM_INV
#define SYN     0.8f    // 1 - DT * TAU_SYN_INV
#define VTH_ENC 0.25f
#define VTH_LIF 0.1f

__device__ __forceinline__ unsigned short f2bf(float f) {
  unsigned u = __builtin_bit_cast(unsigned, f);
  unsigned r = u + 0x7FFFu + ((u >> 16) & 1u);   // round-to-nearest-even
  return (unsigned short)(r >> 16);
}

// LDS byte offset of a __shared__ object (generic AMDGCN LDS pointer keeps the
// wave-relative LDS offset in its low 32 bits).
__device__ __forceinline__ unsigned lds_off(const void* p) {
  return (unsigned)(unsigned long long)p;
}

// Async DMA: global (saddr + 32-bit voffset) -> LDS[ldsaddr], 16 bytes/lane.
// Tracked by ASYNCcnt (s_wait_asynccnt), NOT loadcnt/dscnt.
__device__ __forceinline__ void async_g2l_b128(unsigned ldsaddr, unsigned goff,
                                               unsigned long long sbase) {
  asm volatile("global_load_async_to_lds_b128 %0, %1, %2"
               :: "v"(ldsaddr), "v"(goff), "s"(sbase)
               : "memory");
}

__device__ __forceinline__ void wait_async0() {
  asm volatile("s_wait_asynccnt 0" ::: "memory");
}

// ---------------------------------------------------------------------------
// Encoder: v is a pure function of x (starts at 0, resets on spike), so the
// whole 4-step spike trajectory is computed in ONE pass over x -> 4 u8 planes.
// ---------------------------------------------------------------------------
__global__ void encode_kernel(const float* __restrict__ x,
                              unsigned char* __restrict__ z, size_t n) {
  size_t i = (size_t)blockIdx.x * blockDim.x + threadIdx.x;
  if (i >= n) return;
  float xv = x[i];
  float v = 0.f;
  #pragma unroll
  for (int t = 0; t < 4; ++t) {
    v += DTM * (xv - v);
    unsigned char s = (v - VTH_ENC) > 0.f ? (unsigned char)1 : (unsigned char)0;
    if (s) v = 0.f;
    z[(size_t)t * n + i] = s;
  }
}

// ---------------------------------------------------------------------------
// Weight conversion f32 -> bf16 (optionally zero-padded rows for WMMA tiling)
// ---------------------------------------------------------------------------
__global__ void cvt_bf16(const float* __restrict__ src,
                         unsigned short* __restrict__ dst, size_t n) {
  size_t i = (size_t)blockIdx.x * blockDim.x + threadIdx.x;
  if (i >= n) return;
  dst[i] = f2bf(src[i]);
}

__global__ void cvt_bf16_pad(const float* __restrict__ src,
                             unsigned short* __restrict__ dst,
                             int rows, int padRows, int cols) {
  size_t n = (size_t)padRows * cols;
  size_t i = (size_t)blockIdx.x * blockDim.x + threadIdx.x;
  if (i >= n) return;
  int r = (int)(i / cols);
  int c = (int)(i % cols);
  dst[i] = (r < rows) ? f2bf(src[(size_t)r * cols + c]) : (unsigned short)0;
}

__global__ void zero_f32(float* __restrict__ p, size_t n) {
  size_t i = (size_t)blockIdx.x * blockDim.x + threadIdx.x;
  if (i < n) p[i] = 0.f;
}

// ---------------------------------------------------------------------------
// Main GEMM + fused LIF epilogue.
//   cur = A(MxK, spikes) * Bw(NxK, bf16)^T   via v_wmma_f32_16x16x32_bf16
//   then per element:  v_dec = v + 0.1(i - v); i' = 0.8 i + cur;
//                      s = (v_dec - vth > 0); v' = (1 - s) v_dec
// Block: 256 threads = 8 waves (wave32), 128x128 tile, BK=32.
// Wave grid 4x2; each wave owns 32x64 = 2x4 WMMA 16x16 tiles.
// Double-buffered LDS; bf16 tiles staged with global_load_async_to_lds_b128
// (ASYNCcnt) so the DMA of tile k+1 overlaps the WMMAs of tile k.
// A_U8=true: A is a u8 {0,1} spike plane, splat-converted to bf16 in staging.
// ---------------------------------------------------------------------------
template <bool A_U8>
__global__ __launch_bounds__(256)
void gemm_lif(const void* __restrict__ Aptr,
              const unsigned short* __restrict__ Bw, int K,
              float* __restrict__ V, float* __restrict__ I,
              unsigned short* __restrict__ S, int N)
{
  __shared__ unsigned short lsA[2][128 * 32];
  __shared__ unsigned short lsB[2][128 * 32];

  const int tid   = threadIdx.x;
  const int lane  = tid & 31;
  const int wave  = tid >> 5;
  const int waveM = wave & 3;   // 4 waves along M, 32 rows each
  const int waveN = wave >> 2;  // 2 waves along N, 64 cols each
  const int hh    = lane >> 4;  // half-wave select (K offset 0/8 in frags)
  const int l15   = lane & 15;
  const int bm = blockIdx.x * 128;
  const int bn = blockIdx.y * 128;

  // ---- staging: A tile (128 x 32) ----
  auto stageA = [&](int buf, int k0) {
    if constexpr (A_U8) {
      const unsigned char* A = (const unsigned char*)Aptr;
      #pragma unroll
      for (int i = 0; i < 4; ++i) {
        int idx = tid + i * 256;          // 0..1023 u32 words
        int m   = idx >> 3;
        int kq  = (idx & 7) << 2;         // u8 column
        unsigned z4 = *(const unsigned*)(A + (size_t)(bm + m) * K + k0 + kq);
        // bytes in {0,1} -> packed bf16 {0, 1.0f(0x3F80)}
        unsigned lo = ((z4 & 1u)         * 0x3F80u) |
                      (((z4 >> 8) & 1u)  * 0x3F800000u);
        unsigned hi = (((z4 >> 16) & 1u) * 0x3F80u) |
                      (((z4 >> 24) & 1u) * 0x3F800000u);
        *(unsigned*)&lsA[buf][m * 32 + kq]     = lo;
        *(unsigned*)&lsA[buf][m * 32 + kq + 2] = hi;
      }
    } else {
      #pragma unroll
      for (int i = 0; i < 2; ++i) {
        int idx = tid + i * 256;          // 0..511 uint4 words
        int m   = idx >> 2;
        int kq  = (idx & 3) << 3;
        unsigned goff = (unsigned)(((size_t)(bm + m) * K + k0 + kq) * 2u);
        async_g2l_b128(lds_off(&lsA[buf][m * 32 + kq]), goff,
                       (unsigned long long)Aptr);
      }
    }
  };
  // ---- staging: B tile (128 x 32 of W[n][k], bf16) via async DMA ----
  auto stageB = [&](int buf, int k0) {
    #pragma unroll
    for (int i = 0; i < 2; ++i) {
      int idx = tid + i * 256;
      int n   = idx >> 2;
      int kq  = (idx & 3) << 3;
      unsigned goff = (unsigned)(((size_t)(bn + n) * K + k0 + kq) * 2u);
      async_g2l_b128(lds_off(&lsB[buf][n * 32 + kq]), goff,
                     (unsigned long long)Bw);
    }
  };

  v8f acc[2][4];
  #pragma unroll
  for (int mt = 0; mt < 2; ++mt)
    #pragma unroll
    for (int nt = 0; nt < 4; ++nt)
      #pragma unroll
      for (int r = 0; r < 8; ++r) acc[mt][nt][r] = 0.f;

  stageA(0, 0);
  stageB(0, 0);
  wait_async0();
  __syncthreads();

  int buf = 0;
  for (int k0 = 0; k0 < K; k0 += 32) {
    // issue async DMA of the next tile; it overlaps this tile's WMMAs
    int nk = k0 + 32;
    if (nk < K) { stageA(buf ^ 1, nk); stageB(buf ^ 1, nk); }

    // ---- fragments + 8 WMMAs ----
    FragU fa[2], fb[4];
    #pragma unroll
    for (int mt = 0; mt < 2; ++mt) {
      const unsigned short* pa =
          &lsA[buf][(waveM * 32 + mt * 16 + l15) * 32 + 8 * hh];
      fa[mt].u4[0] = *(const uint4*)pa;          // K = 8h .. 8h+7
      fa[mt].u4[1] = *(const uint4*)(pa + 16);   // K = 16+8h .. 16+8h+7
    }
    #pragma unroll
    for (int nt = 0; nt < 4; ++nt) {
      const unsigned short* pb =
          &lsB[buf][(waveN * 64 + nt * 16 + l15) * 32 + 8 * hh];
      fb[nt].u4[0] = *(const uint4*)pb;
      fb[nt].u4[1] = *(const uint4*)(pb + 16);
    }
    #pragma unroll
    for (int mt = 0; mt < 2; ++mt)
      #pragma unroll
      for (int nt = 0; nt < 4; ++nt)
        acc[mt][nt] = __builtin_amdgcn_wmma_f32_16x16x32_bf16(
            false, fa[mt].v, false, fb[nt].v, (short)0, acc[mt][nt],
            false, false);

    wait_async0();       // next-tile DMA landed in LDS
    __syncthreads();     // all waves done reading cur / writing nxt
    buf ^= 1;
  }

  // ---- fused LIF state update epilogue (C layout: VGPR r -> M=r+8*hh, lane->N) ----
  #pragma unroll
  for (int mt = 0; mt < 2; ++mt) {
    #pragma unroll
    for (int nt = 0; nt < 4; ++nt) {
      int gn = bn + waveN * 64 + nt * 16 + l15;
      #pragma unroll
      for (int r = 0; r < 8; ++r) {
        int gm = bm + waveM * 32 + mt * 16 + r + 8 * hh;
        size_t sidx = (size_t)gm * N + gn;
        float cur  = acc[mt][nt][r];
        float v    = V[sidx];
        float ii   = I[sidx];
        float vdec = v + DTM * (ii - v);
        float idec = SYN * ii;
        float zz   = (vdec - VTH_LIF) > 0.f ? 1.f : 0.f;
        V[sidx] = (1.f - zz) * vdec;
        I[sidx] = idec + cur;
        S[sidx] = (zz != 0.f) ? (unsigned short)0x3F80 : (unsigned short)0;
      }
    }
  }
}

// ---------------------------------------------------------------------------
// Small-N LI heads: one wave per 16x16 output tile, fragments straight from
// global (weights zero-padded to N multiple of 16). Fused LI update; final
// membrane voltage written to d_out every step (last step's value persists).
// ---------------------------------------------------------------------------
__global__ __launch_bounds__(32)
void li_gemm(const unsigned short* __restrict__ S7,
             const unsigned short* __restrict__ Wb, int K,
             float* __restrict__ V, float* __restrict__ I,
             float* __restrict__ out, int Npad, int NC)
{
  const int lane = threadIdx.x & 31;
  const int hh   = lane >> 4;
  const int l15  = lane & 15;
  const int bm = blockIdx.x * 16;
  const int bn = blockIdx.y * 16;

  v8f acc;
  #pragma unroll
  for (int r = 0; r < 8; ++r) acc[r] = 0.f;

  for (int k0 = 0; k0 < K; k0 += 32) {
    FragU fa, fb;
    const unsigned short* pa = S7 + (size_t)(bm + l15) * K + k0 + 8 * hh;
    fa.u4[0] = *(const uint4*)pa;
    fa.u4[1] = *(const uint4*)(pa + 16);
    const unsigned short* pb = Wb + (size_t)(bn + l15) * K + k0 + 8 * hh;
    fb.u4[0] = *(const uint4*)pb;
    fb.u4[1] = *(const uint4*)(pb + 16);
    acc = __builtin_amdgcn_wmma_f32_16x16x32_bf16(
        false, fa.v, false, fb.v, (short)0, acc, false, false);
  }

  int gn = bn + l15;
  #pragma unroll
  for (int r = 0; r < 8; ++r) {
    int gm = bm + r + 8 * hh;
    size_t sidx = (size_t)gm * Npad + gn;
    float cur = acc[r];
    float v   = V[sidx];
    float ii  = I[sidx];
    float vn  = v + DTM * (ii - v);
    float in_ = SYN * ii + cur;
    V[sidx] = vn;
    I[sidx] = in_;
    if (gn < NC) out[(size_t)gm * NC + gn] = vn;
  }
}

// ---------------------------------------------------------------------------
// Launch
// ---------------------------------------------------------------------------
extern "C" void kernel_launch(void* const* d_in, const int* in_sizes, int n_in,
                              void* d_out, int out_size, void* d_ws, size_t ws_size,
                              hipStream_t stream)
{
  const float* x    = (const float*)d_in[0];
  const float* W6   = (const float*)d_in[1];
  const float* W7   = (const float*)d_in[2];
  const float* Wcls = (const float*)d_in[3];
  const float* Wbox = (const float*)d_in[4];

  const int B = 4096, C = 12544, R = 1024;
  const int NCls = 91, NBox = 364, NClsP = 96, NBoxP = 368;
  const size_t nx = (size_t)B * C;

  char* p = (char*)d_ws;
  auto take = [&](size_t bytes) {
    void* r = (void*)p;
    p += (bytes + 255) & ~(size_t)255;
    return r;
  };

  unsigned char*  z      = (unsigned char*) take(4 * nx);                 // 4 spike planes, u8
  unsigned short* w6bf   = (unsigned short*)take((size_t)R * C * 2);
  unsigned short* w7bf   = (unsigned short*)take((size_t)R * R * 2);
  unsigned short* wclsbf = (unsigned short*)take((size_t)NClsP * R * 2);
  unsigned short* wboxbf = (unsigned short*)take((size_t)NBoxP * R * 2);
  unsigned short* s6     = (unsigned short*)take((size_t)B * R * 2);
  unsigned short* s7     = (unsigned short*)take((size_t)B * R * 2);

  size_t stateN = (size_t)4 * B * R + 2 * (size_t)B * NClsP + 2 * (size_t)B * NBoxP;
  float* state = (float*)take(stateN * 4);
  float* v6 = state;
  float* i6 = v6 + (size_t)B * R;
  float* v7 = i6 + (size_t)B * R;
  float* i7 = v7 + (size_t)B * R;
  float* vc = i7 + (size_t)B * R;
  float* ic = vc + (size_t)B * NClsP;
  float* vb = ic + (size_t)B * NClsP;
  float* ib = vb + (size_t)B * NBoxP;
  (void)ib; (void)in_sizes; (void)n_in; (void)out_size; (void)ws_size;

  // Weight conversions (deterministic, every call)
  {
    size_t n = (size_t)R * C;
    cvt_bf16<<<dim3((unsigned)((n + 255) / 256)), 256, 0, stream>>>(W6, w6bf, n);
    n = (size_t)R * R;
    cvt_bf16<<<dim3((unsigned)((n + 255) / 256)), 256, 0, stream>>>(W7, w7bf, n);
    n = (size_t)NClsP * R;
    cvt_bf16_pad<<<dim3((unsigned)((n + 255) / 256)), 256, 0, stream>>>(Wcls, wclsbf, NCls, NClsP, R);
    n = (size_t)NBoxP * R;
    cvt_bf16_pad<<<dim3((unsigned)((n + 255) / 256)), 256, 0, stream>>>(Wbox, wboxbf, NBox, NBoxP, R);
  }

  // Zero all neuron state
  zero_f32<<<dim3((unsigned)((stateN + 255) / 256)), 256, 0, stream>>>(state, stateN);

  // Encoder: single pass over x -> 4 u8 spike planes
  encode_kernel<<<dim3((unsigned)((nx + 255) / 256)), 256, 0, stream>>>(x, z, nx);

  float* out_cls = (float*)d_out;
  float* out_box = out_cls + (size_t)B * NCls;

  for (int t = 0; t < 4; ++t) {
    // layer 6: z_t @ W6^T  + fused LIF -> s6
    gemm_lif<true><<<dim3(B / 128, R / 128), 256, 0, stream>>>(
        (const void*)(z + (size_t)t * nx), w6bf, C, v6, i6, s6, R);
    // layer 7: s6 @ W7^T + fused LIF -> s7
    gemm_lif<false><<<dim3(B / 128, R / 128), 256, 0, stream>>>(
        (const void*)s6, w7bf, R, v7, i7, s7, R);
    // LI heads: write final membrane voltages directly to d_out
    li_gemm<<<dim3(B / 16, NClsP / 16), 32, 0, stream>>>(
        s7, wclsbf, R, vc, ic, out_cls, NClsP, NCls);
    li_gemm<<<dim3(B / 16, NBoxP / 16), 32, 0, stream>>>(
        s7, wboxbf, R, vb, ib, out_box, NBoxP, NBox);
  }
}
